// CrossAttention_47811575939396
// MI455X (gfx1250) — compile-verified, matrix operand order checked by reading, not run on hardware
//
#include <hip/hip_runtime.h>
#include <hip/hip_bf16.h>

// MI455X / gfx1250, wave32. GEMM via v_wmma_f32_16x16x32_bf16, double-buffered
// LDS fed by global_load_async_to_lds_b128 (ASYNCcnt) for A and the Tensor
// Data Mover (tensor_load_to_lds, TENSORcnt) for B.
typedef __bf16 bf16_t;
typedef __attribute__((ext_vector_type(16))) __bf16 v16bf;
typedef __attribute__((ext_vector_type(8)))  __bf16 v8bf;
typedef __attribute__((ext_vector_type(8)))  float  v8f;
typedef __attribute__((ext_vector_type(4)))  unsigned int u32x4;
typedef __attribute__((ext_vector_type(4)))  int i32x4;
typedef __attribute__((ext_vector_type(8)))  int i32x8;

#define LDTE 72  // LDS row stride in bf16 (64 data + 8 pad) -> 144B rows

__device__ __forceinline__ float warp_sum(float v) {
#pragma unroll
  for (int m = 16; m >= 1; m >>= 1) v += __shfl_xor(v, m, 32);
  return v;
}
__device__ __forceinline__ float warp_max(float v) {
#pragma unroll
  for (int m = 16; m >= 1; m >>= 1) v = fmaxf(v, __shfl_xor(v, m, 32));
  return v;
}

// ---------------------------------------------------------------------------
// Async global->LDS staging (no transpose): 128 rows x 64 k bf16 tile.
// 256 threads x 4 x 16B async copies -> ASYNCcnt += 4 per wave per tile.
// ---------------------------------------------------------------------------
__device__ __forceinline__ void stage_async_notrans(unsigned lds_base,
                                                    const bf16_t* gbase, int ld,
                                                    int rc0, int k0, int tid) {
#pragma unroll
  for (int r = 0; r < 4; ++r) {
    const int idx = r * 256 + tid;
    const int row = idx >> 3, kc = (idx & 7) * 8;
    const unsigned loff = lds_base + (unsigned)(row * LDTE + kc) * 2u;
    const unsigned long long ga = (unsigned long long)(uintptr_t)(
        gbase + (long long)(rc0 + row) * ld + (k0 + kc));
    asm volatile("global_load_async_to_lds_b128 %0, %1, off"
                 :
                 : "v"(loff), "v"(ga)
                 : "memory");
  }
}

// ---------------------------------------------------------------------------
// Tensor Data Mover staging of a 128(row) x 64(k) bf16 tile, one instruction
// per tile (issued by wave 0 only). D# packed per CDNA5 ISA ch.8:
//   group0: count=1 | lds_addr | global_addr(57b) | type=2
//   group1: data_size=1(2B), pad_enable, pad_interval=4 (32 DW) ,
//           pad_amount=3 (4 DW)  -> reproduces the LDTE=64+8 LDS layout,
//           tensor_dim0=64, tensor_dim1=128, tile_dim0=64, tile_dim1=128,
//           tensor_dim0_stride=ld
// ---------------------------------------------------------------------------
__device__ __forceinline__ void tdm_load_tile(unsigned lds_byte_addr,
                                              const bf16_t* gtile, int ld) {
  const unsigned long long ga = (unsigned long long)(uintptr_t)gtile;
  u32x4 g0;
  g0[0] = 1u;  // count=1, user descriptor, no gather
  g0[1] = lds_byte_addr;
  g0[2] = (unsigned)(ga & 0xFFFFFFFFu);
  g0[3] = (unsigned)((ga >> 32) & 0x01FFFFFFu) | (2u << 30);  // type=2 (image)
  i32x8 g1;
  g1[0] = (int)((1u << 16) |   // data_size = 2 bytes
                (1u << 20) |   // pad_enable
                (4u << 22) |   // pad_interval: 32 DWORDs
                (3u << 25));   // pad_amount: 4 DWORDs
  g1[1] = (int)((64u & 0xFFFFu) << 16);                   // tensor_dim0 lo16
  g1[2] = (int)((64u >> 16) | ((128u & 0xFFFFu) << 16));  // dim0 hi, dim1 lo
  g1[3] = (int)((128u >> 16) | (64u << 16));              // dim1 hi, tile_dim0
  g1[4] = (int)128u;                                      // tile_dim1 (dim2=0)
  g1[5] = (int)(unsigned)ld;                              // dim0_stride lo32
  g1[6] = 0;                                              // stride hi, d1s lo
  g1[7] = 0;
  i32x4 gz;
  gz[0] = gz[1] = gz[2] = gz[3] = 0;
#if __clang_major__ >= 23
  i32x8 gz8;
#pragma unroll
  for (int i = 0; i < 8; ++i) gz8[i] = 0;
  __builtin_amdgcn_tensor_load_to_lds(g0, g1, gz, gz, gz8, 0);
#else
  __builtin_amdgcn_tensor_load_to_lds(g0, g1, gz, gz, 0);
#endif
}

// Transpose staging: source stored [k][c]; dst [c][k]. Register-buffered.
__device__ __forceinline__ void stage_trans(bf16_t* dst, const bf16_t* gbase,
                                            int ld, int c0, int k0, int tid) {
  v8bf tmp[4];
  int mcv[4], krv[4];
#pragma unroll
  for (int r = 0; r < 4; ++r) {
    const int idx = r * 256 + tid;
    krv[r] = idx >> 4;
    mcv[r] = (idx & 15) * 8;
    const bf16_t* g = gbase + (long long)(k0 + krv[r]) * ld + (c0 + mcv[r]);
    tmp[r] = *(const v8bf*)g;
    __builtin_prefetch(g + 64LL * ld, 0, 1);
  }
#pragma unroll
  for (int r = 0; r < 4; ++r)
#pragma unroll
    for (int j = 0; j < 8; ++j) dst[(mcv[r] + j) * LDTE + krv[r]] = tmp[r][j];
}

template <int N>
__device__ __forceinline__ void wait_async() {
  asm volatile("s_wait_asynccnt %0" : : "n"(N) : "memory");
}

// Exact 16-bit WMMA operand layout: low lanes K chunks {0..7,16..23},
// high lanes {8..15,24..31}; two 16B LDS reads per fragment.
__device__ __forceinline__ v16bf load_frag(const bf16_t* p, int koff) {
  v8bf lo = *(const v8bf*)(p + koff);
  v8bf hi = *(const v8bf*)(p + koff + 16);
  return __builtin_shufflevector(lo, hi, 0, 1, 2, 3, 4, 5, 6, 7, 8, 9, 10, 11,
                                 12, 13, 14, 15);
}

// ---------------------------------------------------------------------------
// LayerNorm (f32 in) -> bf16 out for both tensors. One wave32 per row (D=1024).
// ---------------------------------------------------------------------------
__global__ __launch_bounds__(256) void ln_to_bf16_kernel(
    const float* __restrict__ img, const float* __restrict__ txt,
    const float* __restrict__ w, const float* __restrict__ b,
    bf16_t* __restrict__ nimg, bf16_t* __restrict__ ntxt, int rowsPerTensor) {
  const int D = 1024;
  const int wave = threadIdx.x >> 5, lane = threadIdx.x & 31;
  const int row = blockIdx.x * 8 + wave;
  const float* src;
  bf16_t* dst;
  if (row < rowsPerTensor) {
    src = img + (size_t)row * D;
    dst = nimg + (size_t)row * D;
  } else {
    const int r = row - rowsPerTensor;
    src = txt + (size_t)r * D;
    dst = ntxt + (size_t)r * D;
  }

  float4 x[8];
  float s = 0.f, s2 = 0.f;
#pragma unroll
  for (int i = 0; i < 8; ++i) {
    x[i] = *(const float4*)(src + (i * 32 + lane) * 4);
    s += x[i].x + x[i].y + x[i].z + x[i].w;
    s2 += x[i].x * x[i].x + x[i].y * x[i].y + x[i].z * x[i].z + x[i].w * x[i].w;
  }
  s = warp_sum(s);
  s2 = warp_sum(s2);
  const float mu = s * (1.f / 1024.f);
  const float var = s2 * (1.f / 1024.f) - mu * mu;
  const float rs = rsqrtf(var + 1e-5f);
#pragma unroll
  for (int i = 0; i < 8; ++i) {
    const int d0 = (i * 32 + lane) * 4;
    const float4 wv = *(const float4*)(w + d0);
    const float4 bv = *(const float4*)(b + d0);
    dst[d0 + 0] = (bf16_t)((x[i].x - mu) * rs * wv.x + bv.x);
    dst[d0 + 1] = (bf16_t)((x[i].y - mu) * rs * wv.y + bv.y);
    dst[d0 + 2] = (bf16_t)((x[i].z - mu) * rs * wv.z + bv.z);
    dst[d0 + 3] = (bf16_t)((x[i].w - mu) * rs * wv.w + bv.w);
  }
}

// ---------------------------------------------------------------------------
// Generic bf16 WMMA GEMM: C[m][n] = alpha * sum_k Aeff[m][k]*Beff[n][k] (+bias)
//   TA=false: Aeff[m][k] at A + m*lda + k     TA=true: at A + k*lda + m
//   TB=false: Beff[n][k] at B + n*ldb + k     TB=true: at B + k*ldb + n
// 128x128 tile, 256 threads (8 waves), wave = 32x64 sub-tile, K-step 64,
// double-buffered LDS. A: async copies; B: TDM; transposed: VALU staging.
// M,N multiples of 128; K multiple of 64, K/64 >= 2 (true for all call sites).
// ---------------------------------------------------------------------------
template <bool TA, bool TB, bool OUT_BF16, bool HAS_BIAS>
__global__ __launch_bounds__(256) void gemm_wmma_kernel(
    const bf16_t* __restrict__ A, long long sA, int lda,
    const bf16_t* __restrict__ B, long long sB, int ldb, void* __restrict__ Cv,
    long long sC, int ldc, const float* __restrict__ bias, float alpha, int K) {
  __shared__ alignas(16) bf16_t smem[2][2][128 * LDTE];

  const int tid = threadIdx.x;
  const int m0 = blockIdx.y * 128, n0 = blockIdx.x * 128;
  A += (long long)blockIdx.z * sA;
  B += (long long)blockIdx.z * sB;
  float* Cf = (float*)Cv + (long long)blockIdx.z * sC;
  bf16_t* Cb = (bf16_t*)Cv + (long long)blockIdx.z * sC;

  const int wave = tid >> 5, lane = tid & 31;
  const int wr = wave & 3, wc = wave >> 2;  // 4 row groups x 2 col groups
  const int l16 = lane & 15;
  const int koff = (lane & 16) ? 8 : 0;

  v8f acc[2][4];
#pragma unroll
  for (int mt = 0; mt < 2; ++mt)
#pragma unroll
    for (int nt = 0; nt < 4; ++nt)
#pragma unroll
      for (int i = 0; i < 8; ++i) acc[mt][nt][i] = 0.f;

  unsigned ldsA[2], ldsB[2];
  ldsA[0] = (unsigned)(uintptr_t)&smem[0][0][0];
  ldsB[0] = (unsigned)(uintptr_t)&smem[0][1][0];
  ldsA[1] = (unsigned)(uintptr_t)&smem[1][0][0];
  ldsB[1] = (unsigned)(uintptr_t)&smem[1][1][0];

  auto stage = [&](int buf, int k0) {
    if constexpr (!TA)
      stage_async_notrans(ldsA[buf], A, lda, m0, k0, tid);
    else
      stage_trans(&smem[buf][0][0], A, lda, m0, k0, tid);
    if constexpr (!TB) {
      if (wave == 0)
        tdm_load_tile(ldsB[buf], B + (long long)n0 * ldb + k0, ldb);
    } else {
      stage_trans(&smem[buf][1][0], B, ldb, n0, k0, tid);
    }
  };

  auto compute = [&](int buf) {
    const bf16_t* As = &smem[buf][0][0];
    const bf16_t* Bs = &smem[buf][1][0];
#pragma unroll
    for (int h = 0; h < 2; ++h) {
      const int kb = h * 32;
      v16bf af[2], bfr[4];
#pragma unroll
      for (int mt = 0; mt < 2; ++mt)
        af[mt] = load_frag(As + (wr * 32 + mt * 16 + l16) * LDTE + kb, koff);
#pragma unroll
      for (int nt = 0; nt < 4; ++nt)
        bfr[nt] = load_frag(Bs + (wc * 64 + nt * 16 + l16) * LDTE + kb, koff);
#pragma unroll
      for (int mt = 0; mt < 2; ++mt)
#pragma unroll
        for (int nt = 0; nt < 4; ++nt)
          acc[mt][nt] = __builtin_amdgcn_wmma_f32_16x16x32_bf16(
              false, af[mt], false, bfr[nt], (short)0, acc[mt][nt], false,
              false);
    }
  };

  const int nk = K / 64;  // >= 2
  stage(0, 0);            // prologue: tile 0 -> buffer 0

  // steady state (peeled epilogue keeps the loop branch-free)
  for (int i = 0; i < nk - 1; ++i) {
    const int cur = i & 1;
    stage(cur ^ 1, (i + 1) * 64);  // safe: trailing barrier of iter i-1
    if constexpr (!TA) wait_async<4>();  // current A copies landed
    if constexpr (!TB) {
      if (wave == 0) __builtin_amdgcn_s_wait_tensorcnt((short)1);
    }
    __syncthreads();
    compute(cur);
    __syncthreads();
  }
  // last tile
  if constexpr (!TA) wait_async<0>();
  if constexpr (!TB) {
    if (wave == 0) __builtin_amdgcn_s_wait_tensorcnt((short)0);
  }
  __syncthreads();
  compute((nk - 1) & 1);

  // epilogue: C/D layout N=lane[0:15], VGPR r -> M=r (+8 for hi lanes)
  const int half8 = (lane & 16) ? 8 : 0;
#pragma unroll
  for (int mt = 0; mt < 2; ++mt)
#pragma unroll
    for (int nt = 0; nt < 4; ++nt) {
      const int n = n0 + wc * 64 + nt * 16 + l16;
      const float bb = HAS_BIAS ? bias[n] : 0.f;
#pragma unroll
      for (int r = 0; r < 8; ++r) {
        const int m = m0 + wr * 32 + mt * 16 + r + half8;
        const float v = acc[mt][nt][r] * alpha + bb;
        if (OUT_BF16)
          Cb[(long long)m * ldc + n] = (bf16_t)v;
        else
          Cf[(long long)m * ldc + n] = v;
      }
    }
}

// ---------------------------------------------------------------------------
// Row softmax (f32 scores -> bf16 attn). One wave32 per row of T=2048.
// ---------------------------------------------------------------------------
__global__ __launch_bounds__(256) void softmax_bf16_kernel(
    const float* __restrict__ scores, bf16_t* __restrict__ attn, int T) {
  const int wave = threadIdx.x >> 5, lane = threadIdx.x & 31;
  const long long row = (long long)blockIdx.x * 8 + wave;
  const float* s = scores + row * T;
  bf16_t* d = attn + row * T;

  float4 v[16];
  float mx = -3.0e38f;
#pragma unroll
  for (int i = 0; i < 16; ++i) {
    v[i] = *(const float4*)(s + (i * 32 + lane) * 4);
    mx = fmaxf(mx, fmaxf(fmaxf(v[i].x, v[i].y), fmaxf(v[i].z, v[i].w)));
  }
  mx = warp_max(mx);
  float sum = 0.f;
#pragma unroll
  for (int i = 0; i < 16; ++i) {
    v[i].x = __expf(v[i].x - mx);
    v[i].y = __expf(v[i].y - mx);
    v[i].z = __expf(v[i].z - mx);
    v[i].w = __expf(v[i].w - mx);
    sum += v[i].x + v[i].y + v[i].z + v[i].w;
  }
  sum = warp_sum(sum);
  const float inv = 1.f / sum;
#pragma unroll
  for (int i = 0; i < 16; ++i) {
    const int c = (i * 32 + lane) * 4;
    d[c + 0] = (bf16_t)(v[i].x * inv);
    d[c + 1] = (bf16_t)(v[i].y * inv);
    d[c + 2] = (bf16_t)(v[i].z * inv);
    d[c + 3] = (bf16_t)(v[i].w * inv);
  }
}

// f32 -> bf16 conversion (weights).
__global__ __launch_bounds__(256) void f32_to_bf16_kernel(
    const float* __restrict__ src, bf16_t* __restrict__ dst, long long n) {
  const long long i = ((long long)blockIdx.x * 256 + threadIdx.x) * 4;
  if (i + 3 < n) {
    const float4 v = *(const float4*)(src + i);
    dst[i + 0] = (bf16_t)v.x;
    dst[i + 1] = (bf16_t)v.y;
    dst[i + 2] = (bf16_t)v.z;
    dst[i + 3] = (bf16_t)v.w;
  }
}

// ---------------------------------------------------------------------------
// Orchestration. Workspace layout (aliased, ~260 MiB):
//   [0,   64MiB) : bf16 LN outputs (dead after projections)
//   [0,  128MiB) : f32 attn scores (written after LN consumed)
//   [128, 192MiB): bf16 proj_img, proj_txt
//   [192, 256MiB): bf16 attn (post-softmax); bf16 weights just above
// ---------------------------------------------------------------------------
extern "C" void kernel_launch(void* const* d_in, const int* in_sizes, int n_in,
                              void* d_out, int out_size, void* d_ws,
                              size_t ws_size, hipStream_t stream) {
  (void)in_sizes;
  (void)n_in;
  (void)out_size;
  (void)ws_size;
  const int Bn = 8, S = 2048, D = 1024;
  const long long BS = (long long)Bn * S;  // 16384
  const long long ND = BS * D;             // 16,777,216 elems

  const float* img = (const float*)d_in[0];
  const float* txt = (const float*)d_in[1];
  const float* lnw = (const float*)d_in[2];
  const float* lnb = (const float*)d_in[3];
  const float* Wimg = (const float*)d_in[4];
  const float* bimg = (const float*)d_in[5];
  const float* Wtxt = (const float*)d_in[6];
  const float* btxt = (const float*)d_in[7];

  char* ws = (char*)d_ws;
  bf16_t* norm_img = (bf16_t*)ws;
  bf16_t* norm_txt = norm_img + ND;
  float* scores = (float*)ws;  // aliases LN region (dead by then)
  bf16_t* proj_img = (bf16_t*)(ws + (size_t)Bn * S * S * sizeof(float));
  bf16_t* proj_txt = proj_img + ND;
  bf16_t* attn_bf = proj_txt + ND;
  bf16_t* Wimg_bf = attn_bf + BS * S;
  bf16_t* Wtxt_bf = Wimg_bf + (long long)D * D;

  // 0) LayerNorm both tensors -> bf16
  ln_to_bf16_kernel<<<dim3((unsigned)(2 * BS / 8)), 256, 0, stream>>>(
      img, txt, lnw, lnb, norm_img, norm_txt, (int)BS);

  // 0b) weights f32 -> bf16
  f32_to_bf16_kernel<<<dim3((unsigned)((D * (long long)D) / 1024)), 256, 0,
                       stream>>>(Wimg, Wimg_bf, (long long)D * D);
  f32_to_bf16_kernel<<<dim3((unsigned)((D * (long long)D) / 1024)), 256, 0,
                       stream>>>(Wtxt, Wtxt_bf, (long long)D * D);

  // 1) projections: proj = norm @ W^T + b  (NT GEMM, bf16 out)
  gemm_wmma_kernel<false, false, true, true>
      <<<dim3(D / 128, (unsigned)(BS / 128), 1), 256, 0, stream>>>(
          norm_img, 0, D, Wimg_bf, 0, D, (void*)proj_img, 0, D, bimg, 1.0f, D);
  gemm_wmma_kernel<false, false, true, true>
      <<<dim3(D / 128, (unsigned)(BS / 128), 1), 256, 0, stream>>>(
          norm_txt, 0, D, Wtxt_bf, 0, D, (void*)proj_txt, 0, D, btxt, 1.0f, D);

  // 2) scores = (img_proj @ txt_proj^T) * D^-0.5   (f32 out, aliases LN)
  gemm_wmma_kernel<false, false, false, false>
      <<<dim3(S / 128, S / 128, Bn), 256, 0, stream>>>(
          proj_img, (long long)S * D, D, proj_txt, (long long)S * D, D,
          (void*)scores, (long long)S * S, S, nullptr, 0.03125f, D);

  // 3) softmax rows -> bf16 attn
  softmax_bf16_kernel<<<dim3((unsigned)(BS / 8)), 256, 0, stream>>>(scores,
                                                                    attn_bf, S);

  float* out_img = (float*)d_out;
  float* out_txt = out_img + ND;

  // 4) image_out = attn @ txt_proj   (B stored [k=t][n=d] -> TB)
  gemm_wmma_kernel<false, true, false, false>
      <<<dim3(D / 128, S / 128, Bn), 256, 0, stream>>>(
          attn_bf, (long long)S * S, S, proj_txt, (long long)S * D, D,
          (void*)out_img, (long long)S * D, D, nullptr, 1.0f, S);

  // 5) text_out = attn^T @ img_proj  (A stored [k=s][m=t] -> TA; B -> TB)
  gemm_wmma_kernel<true, true, false, false>
      <<<dim3(D / 128, S / 128, Bn), 256, 0, stream>>>(
          attn_bf, (long long)S * S, S, proj_img, (long long)S * D, D,
          (void*)out_txt, (long long)S * D, D, nullptr, 1.0f, S);
}
